// CapsuleLayer_85255100825918
// MI455X (gfx1250) — matrix operand order repaired and checked by reading, not compiled, not run on hardware
//
#include <hip/hip_runtime.h>

// CapsuleLayer dynamic routing, algebraically refactored + fully fused.
//   x: [64,1024,64] f32   weight: [16,64,32] f32   out v: [64,16,32] f32
//
// u_hat is never materialized (it only appears linearly):
//   GEMM1: y = cT x            [16x1024]*[1024x64]   (WMMA, K split over waves)
//   s_c   = y_c W_c            row-replicated-A WMMA  (M=16 dup, N=32, K=64)
//   w~_c  = v_c W_cT           row-replicated-A WMMA  (M=16 dup, N=64, K=32)
//   GEMM2: L^T += w~ x^T       [16x64]*[64x1024]      (WMMA, C-operand accum)
// ~0.7 GFLOP total; HBM traffic = one 16.7 MB read of x (the floor).
//
// One WG per batch b (64 WGs, 512 threads = 16 wave32s). Wave w == capsule.
// Only *strided* loop-invariant fragments are hoisted to VGPRs (x frags for
// GEMM1, W frags for the s-phase); contiguous fragments are rebuilt per use.

typedef __attribute__((ext_vector_type(16))) unsigned short v16u;
typedef __attribute__((ext_vector_type(16))) __bf16         v16bf;
typedef __attribute__((ext_vector_type(8)))  float          v8f;

#define N_IN     1024
#define K_DIM    64
#define D_CAP    32
#define N_CAPS   16
#define ROUTINGS 3
#define THREADS  512
#define NWAVES   16

__device__ __forceinline__ unsigned short f2bf(float f) {
  unsigned int u = __builtin_bit_cast(unsigned int, f);
  u += 0x7FFFu + ((u >> 16) & 1u);          // round-to-nearest-even
  return (unsigned short)(u >> 16);
}
__device__ __forceinline__ float bf2f(unsigned short h) {
  unsigned int u = ((unsigned int)h) << 16;
  return __builtin_bit_cast(float, u);
}

#define WMMA_BF16(A, Bm, C) \
  __builtin_amdgcn_wmma_f32_16x16x32_bf16(false, __builtin_bit_cast(v16bf, A), \
                                          false, __builtin_bit_cast(v16bf, Bm), \
                                          (short)0, C, false, false)

__global__ __launch_bounds__(THREADS, 1)
void capsule_routing_kernel(const float* __restrict__ x,
                            const float* __restrict__ wgt,
                            float* __restrict__ out) {
  __shared__ unsigned short xs    [N_IN * K_DIM];            // 128 KB  x[b] bf16 [n][i]
  __shared__ unsigned short wsm   [N_CAPS * K_DIM * D_CAP];  //  64 KB  W bf16 [c][i][d]
  __shared__ float          logits[N_IN * N_CAPS];           //  64 KB  L [n][c]
  __shared__ unsigned short csm_t [N_CAPS * N_IN];           //  32 KB  softmax(L)^T bf16 [c][n]
  __shared__ float          ysh   [N_CAPS * K_DIM];          //   4 KB  y = cT x  [c][i]
  __shared__ float          vbuf  [N_CAPS * D_CAP];          //   2 KB  v [c][d]
  __shared__ unsigned short wtld  [N_CAPS * K_DIM];          //   2 KB  w~ = W v bf16 [c][i]

  const int tid  = threadIdx.x;
  const int b    = blockIdx.x;
  const int wv   = tid >> 5;        // wave id == capsule
  const int lane = tid & 31;
  const int lh   = lane >> 4;       // lane half
  const int ll   = lane & 15;

  // ---- stage x[b] and W into LDS as bf16 (f32x4 -> bf16x4) ---------------
  const float4* xb = (const float4*)(x + (size_t)b * N_IN * K_DIM);
  for (int i = tid; i < (N_IN * K_DIM) / 4; i += THREADS) {
    float4 f = xb[i];
    ushort4 h4 = { f2bf(f.x), f2bf(f.y), f2bf(f.z), f2bf(f.w) };
    *(ushort4*)&xs[i * 4] = h4;
  }
  const float4* wg = (const float4*)wgt;
  for (int i = tid; i < (N_CAPS * K_DIM * D_CAP) / 4; i += THREADS) {
    float4 f = wg[i];
    ushort4 h4 = { f2bf(f.x), f2bf(f.y), f2bf(f.z), f2bf(f.w) };
    *(ushort4*)&wsm[i * 4] = h4;
  }
  for (int i = tid; i < N_IN * N_CAPS; i += THREADS) logits[i] = 0.0f;
  __syncthreads();

  // ---- hoist loop-invariant STRIDED fragments into VGPRs -----------------
  // GEMM1 B = x [K=n][N=i]; wave wv owns K rows [wv*64, wv*64+64).
  v16u Bx1[4][2];                       // [i-tile nc][k-step]
  #pragma unroll
  for (int nc = 0; nc < 4; ++nc)
    #pragma unroll
    for (int ks = 0; ks < 2; ++ks) {
      const int kB = wv * 64 + ks * 32 + lh * 16;
      #pragma unroll
      for (int e = 0; e < 16; ++e)      // strided: paid exactly once
        Bx1[nc][ks][e] = xs[(kB + e) * K_DIM + nc * 16 + ll];
    }
  // s-phase B = W_c [K=i][N=d] (strided down i; loop-invariant).
  v16u Bws[2][2];                       // [k-step ks][d-tile dt]
  #pragma unroll
  for (int ks = 0; ks < 2; ++ks)
    #pragma unroll
    for (int dt = 0; dt < 2; ++dt) {
      #pragma unroll
      for (int e = 0; e < 16; ++e)
        Bws[ks][dt][e] = wsm[(wv * K_DIM + ks * 32 + lh * 16 + e) * D_CAP + dt * 16 + ll];
    }

  for (int it = 0; it < ROUTINGS; ++it) {
    // ---- softmax over the capsule axis (transposed store); zero y --------
    for (int n = tid; n < N_IN; n += THREADS) {
      float l[N_CAPS];
      float mx = -3.402823e38f;
      #pragma unroll
      for (int c = 0; c < N_CAPS; ++c) { l[c] = logits[n * N_CAPS + c]; mx = fmaxf(mx, l[c]); }
      float sum = 0.f;
      #pragma unroll
      for (int c = 0; c < N_CAPS; ++c) { l[c] = __expf(l[c] - mx); sum += l[c]; }
      const float inv = 1.f / sum;
      #pragma unroll
      for (int c = 0; c < N_CAPS; ++c) csm_t[c * N_IN + n] = f2bf(l[c] * inv);
    }
    for (int i = tid; i < N_CAPS * K_DIM; i += THREADS) ysh[i] = 0.0f;
    __syncthreads();

    // ---- GEMM1: y[c,i] = sum_n c[n,c]*x[n,i]  (M=16,N=64,K=1024) ---------
    {
      v16u Af[2];                       // A = c^T [M=c][K=n], contiguous rows
      #pragma unroll
      for (int ks = 0; ks < 2; ++ks) {
        const int kb = wv * 64 + ks * 32 + lh * 16;
        #pragma unroll
        for (int j = 0; j < 8; ++j) {
          Af[ks][j]     = csm_t[ll * N_IN + kb + j];
          Af[ks][8 + j] = csm_t[ll * N_IN + kb + 8 + j];
        }
      }
      #pragma unroll
      for (int nc = 0; nc < 4; ++nc) {
        v8f acc = {};
        acc = WMMA_BF16(Af[0], Bx1[nc][0], acc);
        acc = WMMA_BF16(Af[1], Bx1[nc][1], acc);
        #pragma unroll
        for (int r = 0; r < 8; ++r)     // D: lane=col i, VGPR r=row c
          atomicAdd(&ysh[(lh * 8 + r) * K_DIM + nc * 16 + ll], acc[r]);
      }
    }
    __syncthreads();

    // ---- s_c = y_c W_c via row-replicated-A WMMA; squash -> v ------------
    {
      v16u Ay[2];                       // A[m][k] = y_c[k] for every row m
      #pragma unroll
      for (int ks = 0; ks < 2; ++ks) {
        const int kb = wv * K_DIM + ks * 32 + lh * 16;
        #pragma unroll
        for (int j = 0; j < 8; ++j) {
          Ay[ks][j]     = f2bf(ysh[kb + j]);
          Ay[ks][8 + j] = f2bf(ysh[kb + 8 + j]);
        }
      }
      float s01[2];
      #pragma unroll
      for (int dt = 0; dt < 2; ++dt) {
        v8f acc = {};
        acc = WMMA_BF16(Ay[0], Bws[0][dt], acc);
        acc = WMMA_BF16(Ay[1], Bws[1][dt], acc);
        s01[dt] = acc[0];               // all rows identical; lane ll = col d
      }
      float nrm2 = s01[0] * s01[0] + s01[1] * s01[1];
      #pragma unroll
      for (int m = 8; m >= 1; m >>= 1) nrm2 += __shfl_xor(nrm2, m, 32);
      const float nrm   = sqrtf(nrm2);
      const float scale = (nrm2 / (1.f + nrm2)) / (nrm + 1e-8f);
      if (lh == 0) {
        vbuf[wv * D_CAP + ll]      = scale * s01[0];
        vbuf[wv * D_CAP + 16 + ll] = scale * s01[1];
      }
    }
    __syncthreads();

    if (it < ROUTINGS - 1) {
      // ---- w~_c = v_c W_c^T via row-replicated-A WMMA ---------------------
      {
        v16u Av;                        // A[m][k] = v_c[k] (K=32, one k-step)
        #pragma unroll
        for (int j = 0; j < 8; ++j) {
          Av[j]     = f2bf(vbuf[wv * D_CAP + lh * 16 + j]);
          Av[8 + j] = f2bf(vbuf[wv * D_CAP + lh * 16 + 8 + j]);
        }
        float wval[4];
        #pragma unroll
        for (int nt = 0; nt < 4; ++nt) {
          v16u Bww;                     // B = W_c^T [K=d][N=i]: contiguous
          #pragma unroll
          for (int e = 0; e < 16; ++e)
            Bww[e] = wsm[(wv * K_DIM + nt * 16 + ll) * D_CAP + lh * 16 + e];
          v8f acc = {};
          acc = WMMA_BF16(Av, Bww, acc);
          wval[nt] = acc[0];            // all rows identical; lane ll = col i
        }
        if (lh == 0) {
          #pragma unroll
          for (int nt = 0; nt < 4; ++nt)
            wtld[wv * K_DIM + nt * 16 + ll] = f2bf(wval[nt]);
        }
      }
      __syncthreads();

      // ---- GEMM2 (transposed): L^T[c,n] += w~[c,:].x[n,:]  ----------------
      v16u Aw[2];                       // A = w~ [M=c][K=i], contiguous
      #pragma unroll
      for (int ks = 0; ks < 2; ++ks) {
        const int kb = ks * 32 + lh * 16;
        #pragma unroll
        for (int j = 0; j < 8; ++j) {
          Aw[ks][j]     = wtld[ll * K_DIM + kb + j];
          Aw[ks][8 + j] = wtld[ll * K_DIM + kb + 8 + j];
        }
      }
      #pragma unroll
      for (int q = 0; q < 4; ++q) {
        const int n0 = (wv + q * NWAVES) * 16;
        const int ro = (n0 + ll) * K_DIM + lh * 16;
        v16u b0, b1;                    // B = x^T [K=i][N=n]: contiguous b128
        #pragma unroll
        for (int e = 0; e < 16; ++e) { b0[e] = xs[ro + e]; b1[e] = xs[ro + 32 + e]; }
        const int off = (n0 + ll) * N_CAPS + lh * 8;   // 8 consecutive f32
        float4 c0 = *(const float4*)&logits[off];
        float4 c1 = *(const float4*)&logits[off + 4];
        v8f acc = { c0.x, c0.y, c0.z, c0.w, c1.x, c1.y, c1.z, c1.w };
        acc = WMMA_BF16(Aw[0], b0, acc);
        acc = WMMA_BF16(Aw[1], b1, acc);
        *(float4*)&logits[off]     = float4{ acc[0], acc[1], acc[2], acc[3] };
        *(float4*)&logits[off + 4] = float4{ acc[4], acc[5], acc[6], acc[7] };
      }
      __syncthreads();
    }
  }

  // ---- emit v[b] : 16 capsules x 32 dims == 512 == THREADS ---------------
  out[(size_t)b * (N_CAPS * D_CAP) + tid] = vbuf[tid];
}

extern "C" void kernel_launch(void* const* d_in, const int* in_sizes, int n_in,
                              void* d_out, int out_size, void* d_ws, size_t ws_size,
                              hipStream_t stream) {
  (void)in_sizes; (void)n_in; (void)out_size; (void)d_ws; (void)ws_size;
  const float* x   = (const float*)d_in[0];   // [64, 1024, 64]
  const float* wgt = (const float*)d_in[1];   // [16, 64, 32]
  float*       out = (float*)d_out;           // [64, 16, 32]
  capsule_routing_kernel<<<dim3(64), dim3(THREADS), 0, stream>>>(x, wgt, out);
}
